// CQAttention_33466385171057
// MI455X (gfx1250) — matrix operand order repaired and checked by reading, not compiled, not run on hardware
//
#include <hip/hip_runtime.h>
#include <hip/hip_bf16.h>

#define B_ 16
#define N_ 1024
#define M_ 512
#define D_ 512

typedef __attribute__((ext_vector_type(16))) __bf16 v16bf;
typedef __attribute__((ext_vector_type(8)))  __bf16 v8bf;
typedef __attribute__((ext_vector_type(8)))  float  v8f;

// ---------------------------------------------------------------------------
// Wave-level reductions (wave32 on gfx1250)
// ---------------------------------------------------------------------------
static __device__ __forceinline__ float wave_max(float v) {
#pragma unroll
  for (int off = 16; off > 0; off >>= 1) v = fmaxf(v, __shfl_xor(v, off, 32));
  return v;
}
static __device__ __forceinline__ float wave_sum(float v) {
#pragma unroll
  for (int off = 16; off > 0; off >>= 1) v += __shfl_xor(v, off, 32);
  return v;
}

// ---------------------------------------------------------------------------
// WMMA fragment loads matching CDNA5 ISA 7.12.2 layouts.
// A (16x32 bf16): lane = hf*16 + r; lanes 0-15 hold K 0-7 then 16-23,
//                 lanes 16-31 hold K 8-15 then 24-31 (row r of the tile).
// B (32x16 bf16): lane = hf*16 + c; same K split as A for column c:
//                 K 8*hf..+8 then 16+8*hf..+8. Source is B^T row-major.
// ---------------------------------------------------------------------------
static __device__ __forceinline__ v16bf load_a_frag(const __bf16* __restrict__ row,
                                                    int k0, int hf) {
  v8bf lo = *(const v8bf*)(row + k0 + 8 * hf);
  v8bf hi = *(const v8bf*)(row + k0 + 16 + 8 * hf);
  v16bf r;
#pragma unroll
  for (int i = 0; i < 8; ++i) { r[i] = lo[i]; r[i + 8] = hi[i]; }
  return r;
}
static __device__ __forceinline__ v16bf load_b_frag(const __bf16* __restrict__ row,
                                                    int k0, int hf) {
  v8bf lo = *(const v8bf*)(row + k0 + 8 * hf);
  v8bf hi = *(const v8bf*)(row + k0 + 16 + 8 * hf);
  v16bf r;
#pragma unroll
  for (int i = 0; i < 8; ++i) { r[i] = lo[i]; r[i + 8] = hi[i]; }
  return r;
}
static __device__ __forceinline__ v8f wmma_bf16(v16bf a, v16bf b, v8f c) {
  return __builtin_amdgcn_wmma_f32_16x16x32_bf16(false, a, false, b, (short)0, c,
                                                 false, false);
}

// ---------------------------------------------------------------------------
// Prep: Cmlu = bf16(C * w4mlu), Qb = bf16(Q), rowC = C.w4C, colQ = Q.w4Q
// One wave per row of D=512 floats.
// ---------------------------------------------------------------------------
__global__ __launch_bounds__(256) void prep_rows(
    const float* __restrict__ C, const float* __restrict__ Q,
    const float* __restrict__ w4C, const float* __restrict__ w4Q,
    const float* __restrict__ w4mlu,
    __bf16* __restrict__ Cmlu, __bf16* __restrict__ Qb,
    float* __restrict__ rowC, float* __restrict__ colQ) {
  long wid = (long)blockIdx.x * 8 + (threadIdx.x >> 5);
  int lane = threadIdx.x & 31;
  if (wid < (long)B_ * N_) {
    const float* src = C + wid * D_;
    __bf16* dst = Cmlu + wid * D_;
    float s = 0.f;
#pragma unroll
    for (int i = 0; i < D_ / 32; ++i) {
      int d = lane + 32 * i;
      float x = src[d];
      dst[d] = (__bf16)(x * w4mlu[d]);
      s += x * w4C[d];
    }
    s = wave_sum(s);
    if (lane == 0) rowC[wid] = s;
  } else {
    long q = wid - (long)B_ * N_;   // [0, B*M)
    const float* src = Q + q * D_;
    __bf16* dst = Qb + q * D_;
    float s = 0.f;
#pragma unroll
    for (int i = 0; i < D_ / 32; ++i) {
      int d = lane + 32 * i;
      float x = src[d];
      dst[d] = (__bf16)x;
      s += x * w4Q[d];
    }
    s = wave_sum(s);
    if (lane == 0) colQ[q] = s;
  }
}

// ---------------------------------------------------------------------------
// LDS-tiled transpose, f32 src [rows][cols] -> bf16 dst [cols][rows], batched.
// ---------------------------------------------------------------------------
__global__ __launch_bounds__(256) void transpose_f32_to_bf16(
    const float* __restrict__ src, __bf16* __restrict__ dst,
    int rows, int cols) {
  __shared__ __bf16 tile[32][33];
  long batchOff = (long)blockIdx.z * rows * cols;
  const float* s = src + batchOff;
  __bf16* d = dst + batchOff;
  int tx = threadIdx.x & 31;
  int ty = threadIdx.x >> 5;           // 0..7
  int x = blockIdx.x * 32 + tx;        // col in src
#pragma unroll
  for (int j = 0; j < 4; ++j) {
    int y = blockIdx.y * 32 + ty + 8 * j;
    tile[ty + 8 * j][tx] = (__bf16)s[(long)y * cols + x];
  }
  __syncthreads();
#pragma unroll
  for (int j = 0; j < 4; ++j) {
    int oc = blockIdx.x * 32 + ty + 8 * j;  // new row (= src col)
    int orr = blockIdx.y * 32 + tx;         // new col (= src row)
    d[(long)oc * rows + orr] = tile[tx][ty + 8 * j];
  }
}

// ---------------------------------------------------------------------------
// Batched bf16 GEMM: D[R,S] = A[R,K] * Bt[S,K]^T  (f32 accumulate via WMMA)
// 8 waves / WG as 4x2; each wave owns a 32x64 tile (2 A-frags x 4 B-frags,
// 8 accumulators) -> WG tile 128(R) x 128(S). 12 b128 loads per 8 WMMAs.
// MODE 0: sim epilogue  (+rowAux[row] +colAux[col] +bias) -> f32 ws
// MODE 1: f32 -> outF (strided, A matrix into output slice 1)
// MODE 2: bf16 -> outB (Bt = S1 @ S2^T)
// MODE 3: f32 * Cf32[row,col] -> outF (fused C * Bv into output slice 3)
// ---------------------------------------------------------------------------
template <int MODE>
__global__ __launch_bounds__(256) void gemm_bf16(
    const __bf16* __restrict__ A, const __bf16* __restrict__ Bt,
    int R, int S, int K,
    const float* __restrict__ rowAux, const float* __restrict__ colAux,
    const float* __restrict__ biasPtr, const float* __restrict__ Cf32,
    float* __restrict__ outF, __bf16* __restrict__ outB,
    int ldo, long outBatchStride, long aStride, long btStride) {
  int bat = blockIdx.z;
  const __bf16* Ab = A + (long)bat * aStride;
  const __bf16* Btb = Bt + (long)bat * btStride;
  int lane = threadIdx.x & 31;
  int wave = threadIdx.x >> 5;
  int wr = wave >> 1;                       // 0..3
  int wc = wave & 1;                        // 0..1
  int tileR = blockIdx.y * 128 + wr * 32;
  int tileC = blockIdx.x * 128 + wc * 64;
  int l16 = lane & 15;
  int hf = lane >> 4;

  v8f acc[2][4] = {};
  const __bf16* Arow0 = Ab + (long)(tileR + l16) * K;
  const __bf16* Arow1 = Arow0 + (long)16 * K;
  const __bf16* Brow0 = Btb + (long)(tileC + l16) * K;
  const __bf16* Brow1 = Brow0 + (long)16 * K;
  const __bf16* Brow2 = Brow1 + (long)16 * K;
  const __bf16* Brow3 = Brow2 + (long)16 * K;

  for (int k0 = 0; k0 < K; k0 += 32) {
    v16bf a0 = load_a_frag(Arow0, k0, hf);
    v16bf a1 = load_a_frag(Arow1, k0, hf);
    v16bf b0 = load_b_frag(Brow0, k0, hf);
    v16bf b1 = load_b_frag(Brow1, k0, hf);
    v16bf b2 = load_b_frag(Brow2, k0, hf);
    v16bf b3 = load_b_frag(Brow3, k0, hf);
    acc[0][0] = wmma_bf16(a0, b0, acc[0][0]);
    acc[0][1] = wmma_bf16(a0, b1, acc[0][1]);
    acc[0][2] = wmma_bf16(a0, b2, acc[0][2]);
    acc[0][3] = wmma_bf16(a0, b3, acc[0][3]);
    acc[1][0] = wmma_bf16(a1, b0, acc[1][0]);
    acc[1][1] = wmma_bf16(a1, b1, acc[1][1]);
    acc[1][2] = wmma_bf16(a1, b2, acc[1][2]);
    acc[1][3] = wmma_bf16(a1, b3, acc[1][3]);
  }

  float bias = (MODE == 0) ? biasPtr[0] : 0.f;
#pragma unroll
  for (int i = 0; i < 2; ++i) {
#pragma unroll
    for (int j = 0; j < 4; ++j) {
      int r0 = tileR + 16 * i;
      int col = tileC + 16 * j + l16;
#pragma unroll
      for (int v = 0; v < 8; ++v) {
        int row = r0 + v + 8 * hf;
        float val = acc[i][j][v];
        if (MODE == 0) {
          val += rowAux[(long)bat * R + row] + colAux[(long)bat * S + col] + bias;
          outF[(long)bat * outBatchStride + (long)row * ldo + col] = val;
        } else if (MODE == 1) {
          outF[(long)bat * outBatchStride + (long)row * ldo + col] = val;
        } else if (MODE == 2) {
          outB[(long)bat * outBatchStride + (long)row * ldo + col] = (__bf16)val;
        } else {
          float c = Cf32[((long)bat * R + row) * S + col];
          outF[(long)bat * outBatchStride + (long)row * ldo + col] = val * c;
        }
      }
    }
  }
}

// ---------------------------------------------------------------------------
// Dual masked softmax over M per row of sim; writes S1 (Qmask) & S2 (Cmask)
// as bf16. One wave per row; 16 elements per lane.
// ---------------------------------------------------------------------------
__global__ __launch_bounds__(256) void softmax_dual(
    const float* __restrict__ sim, const float* __restrict__ Qmask,
    const float* __restrict__ Cmask, __bf16* __restrict__ S1,
    __bf16* __restrict__ S2) {
  const float NEG_INF = -__builtin_inff();
  long row = (long)blockIdx.x * 8 + (threadIdx.x >> 5);  // [0, B*N)
  int lane = threadIdx.x & 31;
  long b = row / N_;
  const float* p = sim + row * (long)M_;
  bool cm = (Cmask[row] != 0.f);

  float v[M_ / 32];
  unsigned qbits = 0u;
  float m1 = NEG_INF, m2 = NEG_INF;
#pragma unroll
  for (int i = 0; i < M_ / 32; ++i) {
    int m = lane + 32 * i;
    float x = p[m];
    v[i] = x;
    bool qm = (Qmask[b * M_ + m] != 0.f);
    if (qm) qbits |= (1u << i);
    float x1 = qm ? x : NEG_INF;
    float x2 = cm ? x : NEG_INF;
    m1 = fmaxf(m1, x1);
    m2 = fmaxf(m2, x2);
  }
  m1 = wave_max(m1);
  m2 = wave_max(m2);
  float s1 = 0.f, s2 = 0.f;
#pragma unroll
  for (int i = 0; i < M_ / 32; ++i) {
    float x1 = (qbits >> i & 1u) ? v[i] : NEG_INF;
    float x2 = cm ? v[i] : NEG_INF;
    s1 += __expf(x1 - m1);
    s2 += __expf(x2 - m2);
  }
  s1 = wave_sum(s1);
  s2 = wave_sum(s2);
  float i1 = 1.f / s1, i2 = 1.f / s2;
#pragma unroll
  for (int i = 0; i < M_ / 32; ++i) {
    int m = lane + 32 * i;
    float x1 = (qbits >> i & 1u) ? v[i] : NEG_INF;
    float x2 = cm ? v[i] : NEG_INF;
    S1[row * (long)M_ + m] = (__bf16)(__expf(x1 - m1) * i1);
    S2[row * (long)M_ + m] = (__bf16)(__expf(x2 - m2) * i2);
  }
}

// ---------------------------------------------------------------------------
// Finalize: out slice0 = C, slice2 = C * A  (A already in slice1). float4.
// ---------------------------------------------------------------------------
__global__ __launch_bounds__(256) void finalize(const float* __restrict__ C,
                                                float* __restrict__ out) {
  long idx = (long)blockIdx.x * 256 + threadIdx.x;  // over B*N*D/4
  long row = idx / (D_ / 4);
  long c4 = idx % (D_ / 4);
  const float4* C4 = (const float4*)(C + row * (long)D_);
  float4 c = C4[c4];
  float* orow = out + row * (long)(4 * D_);
  ((float4*)orow)[c4] = c;
  float4 a = ((const float4*)(orow + D_))[c4];
  float4 ca;
  ca.x = c.x * a.x; ca.y = c.y * a.y; ca.z = c.z * a.z; ca.w = c.w * a.w;
  ((float4*)(orow + 2 * D_))[c4] = ca;
}

// ---------------------------------------------------------------------------
extern "C" void kernel_launch(void* const* d_in, const int* in_sizes, int n_in,
                              void* d_out, int out_size, void* d_ws,
                              size_t ws_size, hipStream_t stream) {
  (void)in_sizes; (void)n_in; (void)out_size; (void)ws_size;
  const float* C     = (const float*)d_in[0];
  const float* Q     = (const float*)d_in[1];
  const float* Cmask = (const float*)d_in[2];
  const float* Qmask = (const float*)d_in[3];
  const float* w4C   = (const float*)d_in[4];
  const float* w4Q   = (const float*)d_in[5];
  const float* w4mlu = (const float*)d_in[6];
  const float* bias  = (const float*)d_in[7];
  float* out = (float*)d_out;

  char* w = (char*)d_ws;
  auto carve = [&](size_t bytes) {
    void* p = (void*)w;
    w += (bytes + 255) & ~(size_t)255;
    return p;
  };
  __bf16* Cmlu = (__bf16*)carve((size_t)B_ * N_ * D_ * 2);
  __bf16* Qb   = (__bf16*)carve((size_t)B_ * M_ * D_ * 2);
  __bf16* Qt   = (__bf16*)carve((size_t)B_ * D_ * M_ * 2);
  __bf16* Ct   = (__bf16*)carve((size_t)B_ * D_ * N_ * 2);
  float*  rowC = (float*)carve((size_t)B_ * N_ * 4);
  float*  colQ = (float*)carve((size_t)B_ * M_ * 4);
  float*  sim  = (float*)carve((size_t)B_ * N_ * M_ * 4);   // 33.5 MB
  __bf16* BtM  = (__bf16*)sim;  // aliases sim (B*N*N*2 == B*N*M*4), sim dead by then
  __bf16* S1   = (__bf16*)carve((size_t)B_ * N_ * M_ * 2);
  __bf16* S2   = (__bf16*)carve((size_t)B_ * N_ * M_ * 2);

  // 1) precompute bf16 operands + row/col bias terms
  prep_rows<<<(B_ * (N_ + M_)) / 8, 256, 0, stream>>>(C, Q, w4C, w4Q, w4mlu,
                                                      Cmlu, Qb, rowC, colQ);
  // 2) transposed bf16 copies for B-operands
  transpose_f32_to_bf16<<<dim3(D_ / 32, M_ / 32, B_), 256, 0, stream>>>(Q, Qt, M_, D_);
  transpose_f32_to_bf16<<<dim3(D_ / 32, N_ / 32, B_), 256, 0, stream>>>(C, Ct, N_, D_);

  // 3) sim = Cmlu @ Qb^T + rowC + colQ + bias   [B,N,M] f32
  gemm_bf16<0><<<dim3(M_ / 128, N_ / 128, B_), 256, 0, stream>>>(
      Cmlu, Qb, N_, M_, D_, rowC, colQ, bias, nullptr, sim, nullptr, M_,
      (long)N_ * M_, (long)N_ * D_, (long)M_ * D_);

  // 4) S1 / S2 dual masked softmax -> bf16
  softmax_dual<<<(B_ * N_) / 8, 256, 0, stream>>>(sim, Qmask, Cmask, S1, S2);

  // 5) BtM = S1 @ S2^T   [B,N,N] bf16  (overwrites sim buffer)
  gemm_bf16<2><<<dim3(N_ / 128, N_ / 128, B_), 256, 0, stream>>>(
      S1, S2, N_, N_, M_, nullptr, nullptr, nullptr, nullptr, nullptr, BtM, N_,
      (long)N_ * N_, (long)N_ * M_, (long)N_ * M_);

  // 6) A = S1 @ Q  -> out slice 1 (f32)
  gemm_bf16<1><<<dim3(D_ / 128, N_ / 128, B_), 256, 0, stream>>>(
      S1, Qt, N_, D_, M_, nullptr, nullptr, nullptr, nullptr, out + D_, nullptr,
      4 * D_, (long)N_ * 4 * D_, (long)N_ * M_, (long)D_ * M_);

  // 7) C * (BtM @ C) -> out slice 3 (f32, fused multiply by C)
  gemm_bf16<3><<<dim3(D_ / 128, N_ / 128, B_), 256, 0, stream>>>(
      BtM, Ct, N_, D_, N_, nullptr, nullptr, nullptr, C, out + 3 * D_, nullptr,
      4 * D_, (long)N_ * 4 * D_, (long)N_ * N_, (long)D_ * N_);

  // 8) slice 0 = C, slice 2 = C * A
  finalize<<<(B_ * N_ * D_ / 4) / 256, 256, 0, stream>>>(C, out);
}
